// ZScan_3393024163946
// MI455X (gfx1250) — matrix operand order repaired and checked by reading, not compiled, not run on hardware
//
#include <hip/hip_runtime.h>
#include <hip/hip_bf16.h>

// ---------------- types for WMMA ----------------
typedef __attribute__((ext_vector_type(16))) __bf16 bf16x16;
typedef __attribute__((ext_vector_type(8)))  float  f32x8;

struct U4 { unsigned int a, b, c, d; } __attribute__((aligned(16)));
union FragAB { bf16x16 v; U4 q[2]; };

// fp32 -> bf16 (round-to-nearest-even-ish), stored as ushort
__device__ __forceinline__ unsigned short f2bf(float f) {
    unsigned int u = __float_as_uint(f);
    unsigned int r = u + 0x7FFFu + ((u >> 16) & 1u);
    return (unsigned short)(r >> 16);
}
__device__ __forceinline__ float bf2f(unsigned short s) {
    return __uint_as_float(((unsigned int)s) << 16);
}

// A fragment: 16x32 bf16, row-major source, ld in elements (mult of 8).
// lanes 0-15: M=lane, K 0-7 (v0-3) & 16-23 (v4-7); lanes 16-31: K 8-15 & 24-31.
__device__ __forceinline__ bf16x16 load_frag_A(const unsigned short* base, int ld,
                                               int m0, int k0, int lane) {
    int half = lane >> 4, m = lane & 15;
    const unsigned short* p = base + (m0 + m) * ld + k0 + half * 8;
    FragAB f;
    f.q[0] = *(const U4*)p;
    f.q[1] = *(const U4*)(p + 16);
    return f.v;
}

// B fragment: 32(K)x16(N) bf16 from TRANSPOSED storage WT[n][k], ld in elements.
// lanes 0-15: K 0-7 & 16-23, lanes 16-31: K 8-15 & 24-31; n = lane&15.
__device__ __forceinline__ bf16x16 load_frag_B(const unsigned short* baseT, int ld,
                                               int n0, int k0, int lane) {
    int half = lane >> 4, n = lane & 15;
    const unsigned short* p = baseT + (n0 + n) * ld + k0 + half * 8;
    FragAB f;
    f.q[0] = *(const U4*)p;
    f.q[1] = *(const U4*)(p + 16);
    return f.v;
}

// ---------------- problem constants ----------------
#define LSEQ   64     // Dz (sequence length)
#define CMOD   64     // D_MODEL
#define DI     128    // D_INNER
#define DS     16     // D_STATE
#define RNK    4      // DT_RANK
#define NSEQ   1024   // N*H*W sequences

// LDS layout (bytes). Time-disjoint regions are aliased so the footprint fits
// in 160 KB -> two blocks per 320 KB WGP (4 waves/SIMD32 for latency hiding).
//   - sYbf aliases sWinT  (W_in^T dead after input GEMM; Ybf written in scan)
//   - sDtr/sB/sC alias sU (pre-conv u dead after conv; written in x-proj)
// All offsets 16B aligned; all WMMA row ld's multiple of 8 elements.
#define LD64   72     // padded ld for 64-col bf16 tiles
#define LD128  136    // padded ld for 128-col bf16 tiles
#define OFF_SEQ    0                         // bf16 [64][72]      9216  (ph 0-1)
#define OFF_WINT   9216                      // bf16 [256][72]    36864  (ph 0-1)
#define OFF_YBF    9216                      // bf16 [64][136]    17408  (ph 4-5)
#define OFF_WOUTT  46080                     // bf16 [64][136]    17408
#define OFF_WXPT   63488                     // bf16 [36][128]     9216
#define OFF_WDT    72704                     // f32  [4][128]      2048
#define OFF_CONVW  74752                     // f32  [128][4]      2048
#define OFF_CONVB  76800                     // f32  [128]          512
#define OFF_BDT    77312                     // f32  [128]          512
#define OFF_U      77824                     // f32  [64][128]    32768  (ph 1-2)
#define OFF_DTR    77824                     // f32  [64][4]       1024  (ph 3-4)
#define OFF_B      78848                     // f32  [64][16]      4096  (ph 3-4)
#define OFF_C      82944                     // f32  [64][16]      4096  (ph 3-4)
#define OFF_UC     110592                    // f32  [64][128]    32768
#define OFF_ZB     143360                    // bf16 [64][128]    16384
#define SMEM_BYTES 159744                    // 156 KB

__global__ void mamba_zscan_kernel(const float* __restrict__ x,
                                   const float* __restrict__ Win,
                                   const float* __restrict__ convw,
                                   const float* __restrict__ convb,
                                   const float* __restrict__ Wxp,
                                   const float* __restrict__ Wdt,
                                   const float* __restrict__ bdt,
                                   const float* __restrict__ Alog,
                                   const float* __restrict__ Dskip,
                                   const float* __restrict__ Wout,
                                   float* __restrict__ out) {
    extern __shared__ char smem[];
    unsigned short* sSeq   = (unsigned short*)(smem + OFF_SEQ);
    unsigned short* sWinT  = (unsigned short*)(smem + OFF_WINT);
    unsigned short* sWoutT = (unsigned short*)(smem + OFF_WOUTT);
    unsigned short* sWxpT  = (unsigned short*)(smem + OFF_WXPT);
    float* sWdt   = (float*)(smem + OFF_WDT);
    float* sConvW = (float*)(smem + OFF_CONVW);
    float* sConvB = (float*)(smem + OFF_CONVB);
    float* sBdt   = (float*)(smem + OFF_BDT);
    float* sU     = (float*)(smem + OFF_U);
    float* sUC    = (float*)(smem + OFF_UC);
    unsigned short* sZb  = (unsigned short*)(smem + OFF_ZB);
    float* sDtr   = (float*)(smem + OFF_DTR);
    float* sB     = (float*)(smem + OFF_B);
    float* sC     = (float*)(smem + OFF_C);
    unsigned short* sYbf = (unsigned short*)(smem + OFF_YBF);

    const int tid  = threadIdx.x;           // 256 threads = 8 waves
    const int b    = blockIdx.x;            // sequence id = h*32 + w
    const int wave = tid >> 5;
    const int lane = tid & 31;

    // ---- Phase 0: stage sequence + weights into LDS (weights -> bf16) ----
    for (int i = tid; i < LSEQ * CMOD; i += 256) {        // seq[l][c] = x[c,l,h,w]
        int l = i >> 6, c = i & 63;
        sSeq[l * LD64 + c] = f2bf(x[c * 65536 + l * 1024 + b]);
    }
    for (int i = tid; i < 256 * 64; i += 256) {           // W_in^T [n][k]
        int n = i >> 6, k = i & 63;
        sWinT[n * LD64 + k] = f2bf(Win[k * 256 + n]);
    }
    for (int i = tid; i < 64 * 128; i += 256) {           // W_out^T [n][k]
        int n = i >> 7, k = i & 127;
        sWoutT[n * LD128 + k] = f2bf(Wout[k * 64 + n]);
    }
    for (int i = tid; i < 36 * 128; i += 256) {           // W_xproj^T [j][d]
        int j = i >> 7, d = i & 127;
        sWxpT[j * 128 + d] = f2bf(Wxp[d * 36 + j]);
    }
    for (int i = tid; i < RNK * DI; i += 256) sWdt[i]   = Wdt[i];
    for (int i = tid; i < DI * 4;   i += 256) sConvW[i] = convw[i];
    if (tid < DI) { sConvB[tid] = convb[tid]; sBdt[tid] = bdt[tid]; }
    __syncthreads();

    // ---- Phase 1: xz = seq(64x64) @ W_in(64x256) via WMMA bf16 ----
    // 4x16 C-tiles of 16x16; 8 tiles per wave; K=64 -> 2 wmma per tile.
    for (int t = 0; t < 8; ++t) {
        int tile = wave * 8 + t;
        int mt = tile >> 4, nt = tile & 15;
        f32x8 acc = {0.f,0.f,0.f,0.f,0.f,0.f,0.f,0.f};
#pragma unroll
        for (int kt = 0; kt < 2; ++kt) {
            bf16x16 af = load_frag_A(sSeq,  LD64, mt * 16, kt * 32, lane);
            bf16x16 bf = load_frag_B(sWinT, LD64, nt * 16, kt * 32, lane);
            acc = __builtin_amdgcn_wmma_f32_16x16x32_bf16(
                      false, af, false, bf, (short)0, acc, false, false);
        }
        int half = lane >> 4, nn = lane & 15;
        int col = nt * 16 + nn;
        if (nt < 8) {           // u part (cols 0..127) -> fp32 for conv
#pragma unroll
            for (int r = 0; r < 8; ++r)
                sU[(mt * 16 + r + 8 * half) * DI + col] = acc[r];
        } else {                // z part (cols 128..255) -> bf16 (gated later)
            int zc = col - 128;
#pragma unroll
            for (int r = 0; r < 8; ++r)
                sZb[(mt * 16 + r + 8 * half) * DI + zc] = f2bf(acc[r]);
        }
    }
    __syncthreads();

    // ---- Phase 2: causal depthwise conv(4) + SiLU -> uc ----
    for (int i = tid; i < LSEQ * DI; i += 256) {
        int l = i >> 7, d = i & 127;
        float acc = sConvB[d];
#pragma unroll
        for (int k = 0; k < 4; ++k) {
            int sl = l + k - 3;
            if (sl >= 0) acc += sU[sl * DI + d] * sConvW[d * 4 + k];
        }
        acc = acc / (1.0f + __expf(-acc));   // SiLU
        sUC[i] = acc;
    }
    __syncthreads();            // also: sU dead from here; sDtr/sB/sC alias it

    // ---- Phase 3: x_dbl = uc @ W_xproj (64x128 @ 128x36), split dtr/B/C ----
    for (int i = tid; i < LSEQ * 36; i += 256) {
        int l = i / 36, j = i - l * 36;
        const float* ucrow = sUC + l * DI;
        const unsigned short* wrow = sWxpT + j * 128;
        float acc = 0.f;
        for (int d = 0; d < DI; ++d) acc += ucrow[d] * bf2f(wrow[d]);
        if (j < RNK)            sDtr[l * RNK + j] = acc;
        else if (j < RNK + DS)  sB[l * DS + (j - RNK)] = acc;
        else                    sC[l * DS + (j - RNK - DS)] = acc;
    }
    __syncthreads();

    // ---- Phase 4: selective scan, states in registers ----
    // thread t owns d = t/2 and 8 of the 16 states (half = t&1).
    // sYbf aliases the (now dead) sWinT region.
    {
        int d = tid >> 1, half = tid & 1, s0 = half * 8;
        float a_[8], h_[8];
#pragma unroll
        for (int j = 0; j < 8; ++j) {
            a_[j] = -__expf(Alog[d * DS + s0 + j]);
            h_[j] = 0.f;
        }
        float dskip = Dskip[d];
        float w0 = sWdt[0 * DI + d], w1 = sWdt[1 * DI + d];
        float w2 = sWdt[2 * DI + d], w3 = sWdt[3 * DI + d];
        float bdtv = sBdt[d];
        for (int l = 0; l < LSEQ; ++l) {
            const float* dtr = sDtr + l * RNK;
            float raw = bdtv + dtr[0] * w0 + dtr[1] * w1 + dtr[2] * w2 + dtr[3] * w3;
            float dtv = (raw > 20.f) ? raw : log1pf(__expf(raw));  // softplus
            float ul  = sUC[l * DI + d];
            const float* Bl = sB + l * DS + s0;
            const float* Cl = sC + l * DS + s0;
            float yp = 0.f;
#pragma unroll
            for (int j = 0; j < 8; ++j) {
                float dA = __expf(dtv * a_[j]);
                h_[j] = dA * h_[j] + (dtv * Bl[j]) * ul;
                yp += h_[j] * Cl[j];
            }
            yp += __shfl_xor(yp, 1, 32);         // reduce the two state-halves
            float yv = yp + ul * dskip;
            float zv = bf2f(sZb[l * DI + d]);
            yv *= zv / (1.0f + __expf(-zv));     // * SiLU(z)
            if (half == 0) sYbf[l * LD128 + d] = f2bf(yv);
        }
    }
    __syncthreads();

    // ---- Phase 5: out = Y(64x128) @ W_out(128x64) via WMMA, scatter to HBM ----
    for (int t = 0; t < 2; ++t) {
        int tile = wave + t * 8;                 // 16 tiles, 2 per wave
        int mt = tile >> 2, nt = tile & 3;
        f32x8 acc = {0.f,0.f,0.f,0.f,0.f,0.f,0.f,0.f};
#pragma unroll
        for (int kt = 0; kt < 4; ++kt) {
            bf16x16 af = load_frag_A(sYbf,   LD128, mt * 16, kt * 32, lane);
            bf16x16 bf = load_frag_B(sWoutT, LD128, nt * 16, kt * 32, lane);
            acc = __builtin_amdgcn_wmma_f32_16x16x32_bf16(
                      false, af, false, bf, (short)0, acc, false, false);
        }
        int half = lane >> 4, nn = lane & 15;
        int cc = nt * 16 + nn;
#pragma unroll
        for (int r = 0; r < 8; ++r) {
            int ll = mt * 16 + r + 8 * half;
            out[cc * 65536 + ll * 1024 + b] = acc[r];   // (N,C,Dz,H,W)
        }
    }
}

extern "C" void kernel_launch(void* const* d_in, const int* in_sizes, int n_in,
                              void* d_out, int out_size, void* d_ws, size_t ws_size,
                              hipStream_t stream) {
    (void)in_sizes; (void)n_in; (void)out_size; (void)d_ws; (void)ws_size;
    const float* x     = (const float*)d_in[0];
    const float* Win   = (const float*)d_in[1];
    const float* convw = (const float*)d_in[2];
    const float* convb = (const float*)d_in[3];
    const float* Wxp   = (const float*)d_in[4];
    const float* Wdt   = (const float*)d_in[5];
    const float* bdt   = (const float*)d_in[6];
    const float* Alog  = (const float*)d_in[7];
    const float* Dskip = (const float*)d_in[8];
    const float* Wout  = (const float*)d_in[9];
    float* out = (float*)d_out;

    mamba_zscan_kernel<<<NSEQ, 256, SMEM_BYTES, stream>>>(
        x, Win, convw, convb, Wxp, Wdt, bdt, Alog, Dskip, Wout, out);
}
